// SwinAttentionBlock_39633958208107
// MI455X (gfx1250) — matrix operand order repaired and checked by reading, not compiled, not run on hardware
//
#include <hip/hip_runtime.h>
#include <hip/hip_bf16.h>

typedef _Float16 h16;
typedef __attribute__((ext_vector_type(16))) _Float16 v16h;
typedef __attribute__((ext_vector_type(8)))  float    v8f;

// ---------------- problem constants ----------------
#define CDIM   128
#define HEADS  4
#define HD     32
#define WS     7
#define NTOK   49           // WS*WS
#define NPAD   64           // padded window rows
#define SHIFT  3
#define HH     112
#define WW     112
#define NB     8
#define LTOT   (HH*WW)      // 12544
#define TTOT   (NB*LTOT)    // 100352
#define NWIN_B 256          // (112/7)^2
#define NWIN   (NB*NWIN_B)  // 2048
#define MLPH   512

// ---------------- workspace offsets (bytes) ----------------
#define OFF_QKVWT  0ull                         //  384*128 f16  =   98304
#define OFF_PROJWT 98304ull                     //  128*128 f16  =   32768
#define OFF_FC1WT  131072ull                    //  512*128 f16  =  131072
#define OFF_FC2WT  262144ull                    //  128*512 f16  =  131072
#define OFF_BIASM  393216ull                    //  4*49*49 f32  =   38416
#define OFF_XW     432128ull                    //  2048*64*128 f16 = 33554432
#define XW_BYTES   33554432ull
#define OFF_ATTN   33986560ull                  //  T*128 f32    = 51380224  (later reused as ytok)
#define OFF_XRES   85366784ull                  //  T*128 f32    = 51380224
#define OFF_H16    136747008ull                 //  T*128 f16    = 25690112

// ---------------- WMMA fragment helpers (wave32, 16x16x32 f16) ----------------
union FragU { v16h v; uint4 q[2]; };

__device__ __forceinline__ v8f wmma_f16(v16h a, v16h b, v8f c) {
    return __builtin_amdgcn_wmma_f32_16x16x32_f16(false, a, false, b, (short)0, c, false, false);
}

// A-matrix 16x32: lane holds row mbase+(lane&15); halfs [kb..kb+7] and [kb+16..kb+23], kb = kbase + (lane>>4)*8
__device__ __forceinline__ v16h ld_afrag(const h16* base, int stride, int mbase, int kbase, int lane) {
    int row = mbase + (lane & 15);
    int kb  = kbase + ((lane >> 4) << 3);
    const uint4* p = reinterpret_cast<const uint4*>(base + row * stride + kb);
    FragU f; f.q[0] = p[0]; f.q[1] = p[2];
    return f.v;
}

// B-matrix 32x16 from transposed (NxK) storage: lane holds col nbase+(lane&15); halfs [kb..kb+7] and [kb+16..kb+23], kb = kbase + (lane>>4)*8
__device__ __forceinline__ v16h ld_bfrag(const h16* baseT, int stride, int nbase, int kbase, int lane) {
    int n  = nbase + (lane & 15);
    int kb = kbase + ((lane >> 4) << 3);
    const uint4* p = reinterpret_cast<const uint4*>(baseT + n * stride + kb);
    FragU f; f.q[0] = p[0]; f.q[1] = p[2];
    return f.v;
}

// ---------------- kernel 0: weight transpose (f32 -> f16 NxK) + bias gather ----------------
__global__ __launch_bounds__(256) void k_prep(
    const float* __restrict__ qkv_w, const float* __restrict__ proj_w,
    const float* __restrict__ fc1_w, const float* __restrict__ fc2_w,
    const float* __restrict__ tbl, const int* __restrict__ ridx,
    h16* qkv_wt, h16* proj_wt, h16* fc1_wt, h16* fc2_wt, float* biasm)
{
    int i = blockIdx.x * 256 + threadIdx.x;
    if (i < 49152) {                                   // qkv_wt[n][k] = qkv_w[k][n]
        int n = i / 128, k = i % 128;
        qkv_wt[i] = (h16)qkv_w[k * 384 + n];
    } else if (i < 65536) {                            // proj
        int j = i - 49152; int n = j / 128, k = j % 128;
        proj_wt[j] = (h16)proj_w[k * 128 + n];
    } else if (i < 131072) {                           // fc1 (512x128)
        int j = i - 65536; int n = j / 128, k = j % 128;
        fc1_wt[j] = (h16)fc1_w[k * 512 + n];
    } else if (i < 196608) {                           // fc2 (128x512)
        int j = i - 131072; int n = j / 512, k = j % 512;
        fc2_wt[j] = (h16)fc2_w[k * 128 + n];
    } else if (i < 206212) {                           // bias[head][m][n]
        int j = i - 196608;
        biasm[j] = tbl[ridx[j % 2401] * 4 + j / 2401];
    }
}

// ---------------- kernel 1: LN1 + shift + window partition -> f16 ----------------
__global__ __launch_bounds__(256) void k_ln1_window(
    const float* __restrict__ x, const float* __restrict__ g, const float* __restrict__ bv,
    h16* __restrict__ xw)
{
    __shared__ float tile[56 * 129];
    __shared__ float smean[56], srstd[56];
    __shared__ int   sdest[56];
    int tid = threadIdx.x;
    int bx = blockIdx.x;
    int w0 = (bx & 1) * 56;
    int h  = (bx >> 1) % HH;
    int b  = bx / (2 * HH);

    for (int idx = tid; idx < 56 * 128; idx += 256) {
        int c = idx / 56, w = idx % 56;
        tile[w * 129 + c] = x[(((size_t)b * CDIM + c) * HH + h) * WW + w0 + w];
    }
    __syncthreads();
    if (tid < 56) {
        float s = 0.f, ss = 0.f;
        for (int c = 0; c < 128; ++c) { float v = tile[tid * 129 + c]; s += v; ss += v * v; }
        float m = s * (1.f / 128.f);
        smean[tid] = m;
        srstd[tid] = rsqrtf(ss * (1.f / 128.f) - m * m + 1e-5f);
        int hs = (h + HH - SHIFT) % HH;
        int ws2 = (w0 + tid + WW - SHIFT) % WW;
        int win = b * NWIN_B + (hs / WS) * 16 + (ws2 / WS);
        int pos = (hs % WS) * WS + (ws2 % WS);
        sdest[tid] = win * NPAD + pos;
    }
    __syncthreads();
    for (int idx = tid; idx < 56 * 128; idx += 256) {
        int w = idx / 128, c = idx % 128;
        float v = (tile[w * 129 + c] - smean[w]) * srstd[w] * g[c] + bv[c];
        xw[(size_t)sdest[w] * CDIM + c] = (h16)v;
    }
}

// ---------------- kernel 2: fused per-window QKV + attention + projection ----------------
// LDS map (dynamic, 119808 B):
//   phase 1-2:  X[64][136] @0 (17408) | W[192][136] @17408 (52224) | QKVo[64][392] @69632 (50176)
//   phase 3-4:  S f32[64][68] @0 | P[64][72] @17408 | VT[128][72] @26624 | O[64][136] @45056
//   phase 5:    PW[128][136] @69632 (reuses QKVo region)
#define ATTN_LDS 119808
__global__ __launch_bounds__(128) void k_attn_win(
    const h16* __restrict__ xw, const h16* __restrict__ qkv_wt, const float* __restrict__ qkv_b,
    const float* __restrict__ biasm, const h16* __restrict__ proj_wt, const float* __restrict__ proj_b,
    float* __restrict__ attn_out)
{
    extern __shared__ __align__(16) char smem[];
    h16*   Xl   = (h16*)(smem);
    h16*   Wl   = (h16*)(smem + 17408);
    h16*   QKVo = (h16*)(smem + 69632);
    float* Sl   = (float*)(smem);
    h16*   Pl   = (h16*)(smem + 17408);
    h16*   VTl  = (h16*)(smem + 26624);
    h16*   Ol   = (h16*)(smem + 45056);
    h16*   PWl  = (h16*)(smem + 69632);

    int tid = threadIdx.x, wave = tid >> 5, lane = tid & 31;
    int win = blockIdx.x;
    int bb = win >> 8, wloc = win & 255;
    int whi = wloc >> 4, wwi = wloc & 15;
    int hi8 = (lane >> 4) << 3;

    // prefetch weights needed later (global_prefetch_b8 path)
    if (tid < 16) {
        __builtin_prefetch(qkv_wt + 192 * CDIM + tid * 64, 0, 1);
        __builtin_prefetch(proj_wt + tid * 64, 0, 1);
    }

    // ---- load X window (64x128 -> stride 136) ----
    {
        const uint4* xg = (const uint4*)(xw + (size_t)win * NPAD * CDIM);
        uint4* xl = (uint4*)Xl;
        for (int idx = tid; idx < 64 * 16; idx += 128) {
            int r = idx >> 4, c = idx & 15;
            xl[r * 17 + c] = xg[r * 16 + c];
        }
    }
    __syncthreads();

    // A-fragments of X are invariant per wave (each wave owns row-tile mt = wave)
    v16h xafr[4];
#pragma unroll
    for (int kk = 0; kk < 4; ++kk) xafr[kk] = ld_afrag(Xl, 136, wave * 16, kk * 32, lane);

    // ---- phase 2: QKV GEMM, weights staged in 2 halves of 192 output cols ----
    for (int whalf = 0; whalf < 2; ++whalf) {
        const uint4* wg = (const uint4*)(qkv_wt + (size_t)whalf * 192 * CDIM);
        uint4* wl = (uint4*)Wl;
        for (int idx = tid; idx < 192 * 16; idx += 128) {
            int r = idx >> 4, c = idx & 15;
            wl[r * 17 + c] = wg[r * 16 + c];
        }
        __syncthreads();
        for (int ntl = 0; ntl < 12; ++ntl) {
            v16h bfr[4];
#pragma unroll
            for (int kk = 0; kk < 4; ++kk) bfr[kk] = ld_bfrag(Wl, 136, ntl * 16, kk * 32, lane);
            v8f acc; for (int i = 0; i < 8; ++i) acc[i] = 0.f;
#pragma unroll
            for (int kk = 0; kk < 4; ++kk) acc = wmma_f16(xafr[kk], bfr[kk], acc);
            int n = whalf * 192 + ntl * 16 + (lane & 15);
            float qb = qkv_b[n];
            float sc = (n < 128) ? 0.17677669529663687f : 1.0f;  // 1/sqrt(hd) folded into Q
#pragma unroll
            for (int r = 0; r < 8; ++r) {
                int m = wave * 16 + r + hi8;
                QKVo[m * 392 + n] = (h16)((acc[r] + qb) * sc);
            }
        }
        __syncthreads();
    }

    // ---- transpose V into VT[d][t] for contiguous B-fragments ----
    for (int idx = tid; idx < 128 * 64; idx += 128) {
        int d = idx >> 6, t = idx & 63;
        VTl[d * 72 + t] = QKVo[t * 392 + 256 + d];
    }
    __syncthreads();

    // ---- phase 4: per-head attention (each wave owns query-row tile mt = wave) ----
    for (int hdi = 0; hdi < HEADS; ++hdi) {
        // S = Q K^T (+bias, -inf padding)
        v16h qaf = ld_afrag(QKVo, 392, wave * 16, hdi * 32, lane);
        for (int nt = 0; nt < 4; ++nt) {
            v16h bf = ld_bfrag(QKVo, 392, nt * 16, 128 + hdi * 32, lane);
            v8f acc; for (int i = 0; i < 8; ++i) acc[i] = 0.f;
            acc = wmma_f16(qaf, bf, acc);
            int n = nt * 16 + (lane & 15);
#pragma unroll
            for (int r = 0; r < 8; ++r) {
                int m = wave * 16 + r + hi8;
                float v = acc[r];
                if (m < NTOK && n < NTOK) v += biasm[hdi * 2401 + m * NTOK + n];
                if (n >= NTOK) v = -1e30f;
                Sl[m * 68 + n] = v;
            }
        }
        __syncthreads();
        // softmax per row -> P (f16)
        if (tid < 64) {
            int m = tid;
            float mx = -1e30f;
            for (int n = 0; n < 64; ++n) mx = fmaxf(mx, Sl[m * 68 + n]);
            float s = 0.f;
            for (int n = 0; n < 64; ++n) { float e = __expf(Sl[m * 68 + n] - mx); Sl[m * 68 + n] = e; s += e; }
            float inv = 1.f / s;
            for (int n = 0; n < 64; ++n) Pl[m * 72 + n] = (h16)(Sl[m * 68 + n] * inv);
        }
        __syncthreads();
        // O = P V
        {
            v16h pafr[2];
#pragma unroll
            for (int kk = 0; kk < 2; ++kk) pafr[kk] = ld_afrag(Pl, 72, wave * 16, kk * 32, lane);
            for (int nt = 0; nt < 2; ++nt) {
                v16h bfr[2];
#pragma unroll
                for (int kk = 0; kk < 2; ++kk) bfr[kk] = ld_bfrag(VTl, 72, hdi * 32 + nt * 16, kk * 32, lane);
                v8f acc; for (int i = 0; i < 8; ++i) acc[i] = 0.f;
#pragma unroll
                for (int kk = 0; kk < 2; ++kk) acc = wmma_f16(pafr[kk], bfr[kk], acc);
                int n = hdi * 32 + nt * 16 + (lane & 15);
#pragma unroll
                for (int r = 0; r < 8; ++r) {
                    int m = wave * 16 + r + hi8;
                    Ol[m * 136 + n] = (h16)acc[r];
                }
            }
        }
        __syncthreads();
    }

    // ---- phase 5: projection + window-reverse scatter ----
    {
        const uint4* pg = (const uint4*)proj_wt;
        uint4* pl = (uint4*)PWl;
        for (int idx = tid; idx < 128 * 16; idx += 128) {
            int r = idx >> 4, c = idx & 15;
            pl[r * 17 + c] = pg[r * 16 + c];
        }
    }
    __syncthreads();
    {
        v16h oafr[4];
#pragma unroll
        for (int kk = 0; kk < 4; ++kk) oafr[kk] = ld_afrag(Ol, 136, wave * 16, kk * 32, lane);
        for (int nt = 0; nt < 8; ++nt) {
            v16h bfr[4];
#pragma unroll
            for (int kk = 0; kk < 4; ++kk) bfr[kk] = ld_bfrag(PWl, 136, nt * 16, kk * 32, lane);
            v8f acc; for (int i = 0; i < 8; ++i) acc[i] = 0.f;
#pragma unroll
            for (int kk = 0; kk < 4; ++kk) acc = wmma_f16(oafr[kk], bfr[kk], acc);
            int n = nt * 16 + (lane & 15);
            float pb = proj_b[n];
#pragma unroll
            for (int r = 0; r < 8; ++r) {
                int m = wave * 16 + r + hi8;
                if (m < NTOK) {
                    int i = m / WS, j = m % WS;
                    int h = (whi * WS + i + SHIFT) % HH;
                    int w = (wwi * WS + j + SHIFT) % WW;
                    attn_out[((size_t)bb * LTOT + h * WW + w) * CDIM + n] = acc[r] + pb;
                }
            }
        }
    }
}

// ---------------- kernel 3: residual add + LN2 -> f16 ----------------
__global__ __launch_bounds__(256) void k_resid_ln2(
    const float* __restrict__ x, const float* __restrict__ attn,
    const float* __restrict__ g2, const float* __restrict__ b2,
    float* __restrict__ xres, h16* __restrict__ h16out)
{
    __shared__ float tile[56 * 129];
    __shared__ float smean[56], srstd[56];
    int tid = threadIdx.x;
    int bx = blockIdx.x;
    int w0 = (bx & 1) * 56;
    int h  = (bx >> 1) % HH;
    int b  = bx / (2 * HH);
    size_t lbase = (size_t)b * LTOT + h * WW + w0;

    for (int idx = tid; idx < 56 * 128; idx += 256) {
        int c = idx / 56, w = idx % 56;
        tile[w * 129 + c] = x[(((size_t)b * CDIM + c) * HH + h) * WW + w0 + w];
    }
    __syncthreads();
    for (int idx = tid; idx < 56 * 128; idx += 256) {
        int w = idx / 128, c = idx % 128;
        size_t gi = (lbase + w) * CDIM + c;
        float v = tile[w * 129 + c] + attn[gi];
        tile[w * 129 + c] = v;
        xres[gi] = v;
    }
    __syncthreads();
    if (tid < 56) {
        float s = 0.f, ss = 0.f;
        for (int c = 0; c < 128; ++c) { float v = tile[tid * 129 + c]; s += v; ss += v * v; }
        float m = s * (1.f / 128.f);
        smean[tid] = m;
        srstd[tid] = rsqrtf(ss * (1.f / 128.f) - m * m + 1e-5f);
    }
    __syncthreads();
    for (int idx = tid; idx < 56 * 128; idx += 256) {
        int w = idx / 128, c = idx % 128;
        float v = (tile[w * 129 + c] - smean[w]) * srstd[w] * g2[c] + b2[c];
        h16out[(lbase + w) * CDIM + c] = (h16)v;
    }
}

// ---------------- kernel 4: fused MLP (fc1 + GELU + fc2 + residual), hmid never leaves LDS ----
// 128-token tile per 256-thread block (8 waves, wave owns row-tile mt = wave).
// LDS map (dynamic, 307200 B <= 320 KB/WGP):
//   AS  @0      : h16 [128][136]  (34816)
//   W   @34816  : h16 [512][136] fc1 weights (139264) ; later fc2 K-chunks [128][136] (34816)
//   MID @174080 : h16 [128][520]  (133120)
#define MLP_LDS 307200
__global__ __launch_bounds__(256) void k_mlp(
    const h16* __restrict__ A, const h16* __restrict__ W1t, const float* __restrict__ b1,
    const h16* __restrict__ W2t, const float* __restrict__ b2,
    const float* __restrict__ xres, float* __restrict__ ytok)
{
    extern __shared__ __align__(16) char smem[];
    h16* As  = (h16*)(smem);
    h16* Wl  = (h16*)(smem + 34816);
    h16* Mid = (h16*)(smem + 174080);

    int tid = threadIdx.x, wave = tid >> 5, lane = tid & 31;
    int m0 = blockIdx.x * 128;
    int hi8 = (lane >> 4) << 3;

    // ---- stage activations (128x128) and the full fc1 weight matrix (512x128) ----
    {
        const uint4* Ag = (const uint4*)A;         // row stride 16 uint4
        uint4* Al = (uint4*)As;
        for (int idx = tid; idx < 128 * 16; idx += 256) {
            int r = idx >> 4, c = idx & 15;
            Al[r * 17 + c] = Ag[(size_t)(m0 + r) * 16 + c];
        }
        const uint4* Wg = (const uint4*)W1t;       // row stride 16 uint4
        uint4* wl = (uint4*)Wl;
        for (int idx = tid; idx < 512 * 16; idx += 256) {
            int r = idx >> 4, c = idx & 15;
            wl[r * 17 + c] = Wg[(size_t)r * 16 + c];
        }
    }
    __syncthreads();

    // ---- fc1 + GELU -> Mid (LDS only) ----
    {
        v16h aafr[4];
#pragma unroll
        for (int kk = 0; kk < 4; ++kk) aafr[kk] = ld_afrag(As, 136, wave * 16, kk * 32, lane);
        for (int nt = 0; nt < 32; ++nt) {
            v16h bfr[4];
#pragma unroll
            for (int kk = 0; kk < 4; ++kk) bfr[kk] = ld_bfrag(Wl, 136, nt * 16, kk * 32, lane);
            v8f acc; for (int i = 0; i < 8; ++i) acc[i] = 0.f;
#pragma unroll
            for (int kk = 0; kk < 4; ++kk) acc = wmma_f16(aafr[kk], bfr[kk], acc);
            int n = nt * 16 + (lane & 15);
            float bn = b1[n];
#pragma unroll
            for (int r = 0; r < 8; ++r) {
                int m = wave * 16 + r + hi8;
                float v = acc[r] + bn;
                v = 0.5f * v * (1.f + erff(v * 0.70710678118654752f));   // exact GELU
                Mid[m * 520 + n] = (h16)v;
            }
        }
    }
    __syncthreads();

    // ---- fc2 (K=512 staged in 4 chunks through the retired fc1-weight region) ----
    v8f acc[8];
    for (int i = 0; i < 8; ++i) for (int j = 0; j < 8; ++j) acc[i][j] = 0.f;
    const uint4* W2g = (const uint4*)W2t;          // row stride 64 uint4
    for (int kb = 0; kb < 4; ++kb) {
        uint4* wl = (uint4*)Wl;
        for (int idx = tid; idx < 128 * 16; idx += 256) {
            int r = idx >> 4, c = idx & 15;
            wl[r * 17 + c] = W2g[(size_t)r * 64 + kb * 16 + c];
        }
        __syncthreads();
        v16h mafr[4];
#pragma unroll
        for (int kk = 0; kk < 4; ++kk) mafr[kk] = ld_afrag(Mid, 520, wave * 16, kb * 128 + kk * 32, lane);
        for (int nt = 0; nt < 8; ++nt) {
            v16h bfr[4];
#pragma unroll
            for (int kk = 0; kk < 4; ++kk) bfr[kk] = ld_bfrag(Wl, 136, nt * 16, kk * 32, lane);
#pragma unroll
            for (int kk = 0; kk < 4; ++kk) acc[nt] = wmma_f16(mafr[kk], bfr[kk], acc[nt]);
        }
        __syncthreads();
    }
    // ---- epilogue: + bias + residual -> token-major f32 ----
    for (int nt = 0; nt < 8; ++nt) {
        int n = nt * 16 + (lane & 15);
        float bn = b2[n];
#pragma unroll
        for (int r = 0; r < 8; ++r) {
            int m = m0 + wave * 16 + r + hi8;
            size_t gi = (size_t)m * CDIM + n;
            ytok[gi] = acc[nt][r] + bn + xres[gi];
        }
    }
}

// ---------------- kernel 5: token-major -> (B,C,H,W) transpose ----------------
__global__ __launch_bounds__(256) void k_out_transpose(
    const float* __restrict__ ytok, float* __restrict__ out)
{
    __shared__ float tile[56 * 129];
    int tid = threadIdx.x;
    int bx = blockIdx.x;
    int w0 = (bx & 1) * 56;
    int h  = (bx >> 1) % HH;
    int b  = bx / (2 * HH);
    size_t lbase = (size_t)b * LTOT + h * WW + w0;

    for (int idx = tid; idx < 56 * 128; idx += 256) {
        int w = idx / 128, c = idx % 128;
        tile[w * 129 + c] = ytok[(lbase + w) * CDIM + c];
    }
    __syncthreads();
    for (int idx = tid; idx < 56 * 128; idx += 256) {
        int c = idx / 56, w = idx % 56;
        out[(((size_t)b * CDIM + c) * HH + h) * WW + w0 + w] = tile[w * 129 + c];
    }
}

// ---------------- host ----------------
extern "C" void kernel_launch(void* const* d_in, const int* in_sizes, int n_in,
                              void* d_out, int out_size, void* d_ws, size_t ws_size,
                              hipStream_t stream) {
    (void)in_sizes; (void)n_in; (void)out_size; (void)ws_size;
    const float* x       = (const float*)d_in[0];
    const float* norm1_g = (const float*)d_in[1];
    const float* norm1_b = (const float*)d_in[2];
    const float* qkv_w   = (const float*)d_in[3];
    const float* qkv_b   = (const float*)d_in[4];
    const float* tbl     = (const float*)d_in[5];
    const int*   ridx    = (const int*)d_in[6];
    const float* proj_w  = (const float*)d_in[7];
    const float* proj_b  = (const float*)d_in[8];
    const float* norm2_g = (const float*)d_in[9];
    const float* norm2_b = (const float*)d_in[10];
    const float* fc1_w   = (const float*)d_in[11];
    const float* fc1_b   = (const float*)d_in[12];
    const float* fc2_w   = (const float*)d_in[13];
    const float* fc2_b   = (const float*)d_in[14];
    float* out = (float*)d_out;

    char* ws = (char*)d_ws;
    h16*   qkv_wt  = (h16*)(ws + OFF_QKVWT);
    h16*   proj_wt = (h16*)(ws + OFF_PROJWT);
    h16*   fc1_wt  = (h16*)(ws + OFF_FC1WT);
    h16*   fc2_wt  = (h16*)(ws + OFF_FC2WT);
    float* biasm   = (float*)(ws + OFF_BIASM);
    h16*   xw      = (h16*)(ws + OFF_XW);
    float* attn    = (float*)(ws + OFF_ATTN);
    float* xres    = (float*)(ws + OFF_XRES);
    h16*   h16buf  = (h16*)(ws + OFF_H16);
    float* ytok    = attn;   // attn buffer free after k_resid_ln2

    hipFuncSetAttribute(reinterpret_cast<const void*>(&k_attn_win),
                        hipFuncAttributeMaxDynamicSharedMemorySize, ATTN_LDS);
    hipFuncSetAttribute(reinterpret_cast<const void*>(&k_mlp),
                        hipFuncAttributeMaxDynamicSharedMemorySize, MLP_LDS);

    k_prep<<<(206212 + 255) / 256, 256, 0, stream>>>(qkv_w, proj_w, fc1_w, fc2_w, tbl, ridx,
                                                     qkv_wt, proj_wt, fc1_wt, fc2_wt, biasm);
    hipMemsetAsync(xw, 0, XW_BYTES, stream);   // zero padded window rows
    k_ln1_window<<<NB * HH * 2, 256, 0, stream>>>(x, norm1_g, norm1_b, xw);
    k_attn_win<<<NWIN, 128, ATTN_LDS, stream>>>(xw, qkv_wt, qkv_b, biasm, proj_wt, proj_b, attn);
    k_resid_ln2<<<NB * HH * 2, 256, 0, stream>>>(x, attn, norm2_g, norm2_b, xres, h16buf);
    k_mlp<<<TTOT / 128, 256, MLP_LDS, stream>>>(h16buf, fc1_wt, fc1_b, fc2_wt, fc2_b, xres, ytok);
    k_out_transpose<<<NB * HH * 2, 256, 0, stream>>>(ytok, out);
}
